// DeepSeekMoE_34840774705582
// MI455X (gfx1250) — matrix operand order repaired and checked by reading, not compiled, not run on hardware
//
#include <hip/hip_runtime.h>
#include <hip/hip_bf16.h>
#include <cstdint>
#include <cstddef>

typedef __attribute__((ext_vector_type(16))) _Float16 v16h;
typedef __attribute__((ext_vector_type(8)))  _Float16 v8h;
typedef __attribute__((ext_vector_type(4)))  _Float16 v4h;
typedef __attribute__((ext_vector_type(8)))  float    v8f;

#define DD    768
#define II    3072
#define EE    8
#define NT    2048
#define TM    64
#define TN    64
#define TKS   32
#define SLIM  10.0f

#define SH_SLOTS      2048                 // shared expert occupies slots [0, 2048): slot == token
#define MAX_EXP_SLOTS 4608                 // sum padded expert slots <= 2*NT + 8*63 = 4600
#define TOT_SLOTS     (SH_SLOTS + MAX_EXP_SLOTS)

// workspace layout (byte offsets, all 16B aligned)
#define OFF_HDR   0                         // int[64]: [0..7]=cnt [8..15]=padded [16..23]=base [24..31]=fill
#define OFF_TOPKI 256                       // int[2*NT]   (expert ids, rewritten to slot ids by fill)
#define OFF_TOPKW (OFF_TOPKI + 2*NT*4)      // float[2*NT]
#define OFF_STOK  (OFF_TOPKW + 2*NT*4)      // int[TOT_SLOTS] slot -> token
#define OFF_H     (OFF_STOK + TOT_SLOTS*4)  // _Float16[TOT_SLOTS*II]
#define OFF_OUT   (OFF_H + (size_t)TOT_SLOTS*II*2) // float[TOT_SLOTS*DD]

__device__ __forceinline__ v8f wmma_f16(v16h a, v16h b, v8f c) {
    return __builtin_amdgcn_wmma_f32_16x16x32_f16(false, a, false, b,
                                                  (short)0, c, false, false);
}

__device__ __forceinline__ v16h cat8(v8h a, v8h b) {
    return __builtin_shufflevector(a, b, 0,1,2,3,4,5,6,7,8,9,10,11,12,13,14,15);
}

__device__ __forceinline__ v8h cvt8(float4 a, float4 b) {
    v8h r;
    r[0] = (_Float16)a.x; r[1] = (_Float16)a.y; r[2] = (_Float16)a.z; r[3] = (_Float16)a.w;
    r[4] = (_Float16)b.x; r[5] = (_Float16)b.y; r[6] = (_Float16)b.z; r[7] = (_Float16)b.w;
    return r;
}

__device__ __forceinline__ v4h cvt4(float a, float b, float c, float d) {
    v4h r;
    r[0] = (_Float16)a; r[1] = (_Float16)b; r[2] = (_Float16)c; r[3] = (_Float16)d;
    return r;
}

// ---------------- init: zero header, slot->token table (shared = identity) ----
__global__ void moe_init(int* __restrict__ hdr, int* __restrict__ stok) {
    int t = blockIdx.x * blockDim.x + threadIdx.x;
    if (t < 64) hdr[t] = 0;
    if (t < TOT_SLOTS) stok[t] = (t < SH_SLOTS) ? t : 0;
}

// ---------------- gating: sqrt(softplus(x@gw)), top-2, normalized weights ----
__global__ __launch_bounds__(256)
void moe_gate(const float* __restrict__ x, const float* __restrict__ gw,
              const float* __restrict__ gb, int* __restrict__ hdr,
              int* __restrict__ tki, float* __restrict__ tkw) {
    __shared__ float gws[DD * EE];
    for (int i = threadIdx.x; i < DD * EE; i += 256) gws[i] = gw[i];
    __syncthreads();

    int t = blockIdx.x * blockDim.x + threadIdx.x;
    if (t >= NT) return;
    const float* xr = x + (size_t)t * DD;
    float acc[EE];
#pragma unroll
    for (int e = 0; e < EE; ++e) acc[e] = 0.f;
    for (int d = 0; d < DD; d += 4) {
        float4 xv = *(const float4*)(xr + d);
#pragma unroll
        for (int e = 0; e < EE; ++e)
            acc[e] += xv.x * gws[(d + 0) * EE + e] + xv.y * gws[(d + 1) * EE + e]
                    + xv.z * gws[(d + 2) * EE + e] + xv.w * gws[(d + 3) * EE + e];
    }
    float sc[EE];
#pragma unroll
    for (int e = 0; e < EE; ++e) {
        float v = acc[e];
        float sp = (v > 20.f) ? v : log1pf(expf(v));   // softplus
        sc[e] = sqrtf(sp);
    }
    int i0 = 0; float b0 = -1e30f;
#pragma unroll
    for (int e = 0; e < EE; ++e) { float r = sc[e] + gb[e]; if (r > b0) { b0 = r; i0 = e; } }
    int i1 = (i0 == 0) ? 1 : 0; float b1 = -1e30f;
#pragma unroll
    for (int e = 0; e < EE; ++e) {
        if (e == i0) continue;
        float r = sc[e] + gb[e]; if (r > b1) { b1 = r; i1 = e; }
    }
    float w0 = sc[i0], w1 = sc[i1];
    float s = fmaxf(w0 + w1, 1e-6f);
    w0 /= s; w1 /= s;
    tki[t * 2] = i0; tki[t * 2 + 1] = i1;
    tkw[t * 2] = w0; tkw[t * 2 + 1] = w1;
    atomicAdd(&hdr[i0], 1);
    atomicAdd(&hdr[i1], 1);
}

// ---------------- scan: padded counts + segment bases (experts after shared) --
__global__ void moe_scan(int* __restrict__ hdr) {
    if (threadIdx.x == 0 && blockIdx.x == 0) {
        int base = SH_SLOTS;
        for (int e = 0; e < EE; ++e) {
            int p = (hdr[e] + TM - 1) / TM * TM;
            hdr[8 + e] = p;
            hdr[16 + e] = base;
            base += p;
        }
    }
}

// ---------------- fill: assign slots; rewrite tki[expert] -> tki[slot] --------
__global__ void moe_fill(int* __restrict__ hdr, int* __restrict__ tki,
                         int* __restrict__ stok) {
    int t = blockIdx.x * blockDim.x + threadIdx.x;
    if (t >= 2 * NT) return;
    int e = tki[t];
    int pos = atomicAdd(&hdr[24 + e], 1);
    int slot = hdr[16 + e] + pos;
    stok[slot] = t >> 1;
    tki[t] = slot;
}

// ---------------- GEMM1: h = silu(min(x@W1,L)) * clip(x@W3,+-L), f16 out ------
// 256 thr = 8 waves; 64x64 tile; double-buffered LDS; global loads for the next
// k-slice issued before the WMMAs; a sched_barrier pins ALL fragment ds_loads
// before the WMMA chain so the wait inserter can emit partial s_wait_dscnt.
__global__ __launch_bounds__(256)
void moe_gemm1(const float* __restrict__ x,
               const float* __restrict__ w1, const float* __restrict__ w3,
               const float* __restrict__ w1s, const float* __restrict__ w3s,
               const int* __restrict__ hdr, const int* __restrict__ stok,
               _Float16* __restrict__ h) {
    int g = blockIdx.z;                       // 0 = shared expert, 1..8 = routed
    int base, cp;
    const float *W1, *W3;
    if (g == 0) { base = 0; cp = SH_SLOTS; W1 = w1s; W3 = w3s; }
    else {
        int e = g - 1;
        base = hdr[16 + e]; cp = hdr[8 + e];
        W1 = w1 + (size_t)e * DD * II;
        W3 = w3 + (size_t)e * DD * II;
    }
    int rowT = blockIdx.y * TM;
    if (rowT >= cp) return;                   // static worst-case grid, early exit
    int colB = blockIdx.x * TN;

    __shared__ __align__(16) _Float16 As [2 * TM * 40];
    __shared__ __align__(16) _Float16 B1s[2 * TN * 40];
    __shared__ __align__(16) _Float16 B3s[2 * TN * 40];

    int tid = threadIdx.x;
    // staging roles: A -> 8-float chunk of one gathered row; B -> 8 k-values of one col
    int arow = tid >> 2, acol = (tid & 3) * 8;
    const float* xrow = x + (size_t)stok[base + rowT + arow] * DD;   // token idx in a register
    int bn = tid & 63, bk = (tid >> 6) * 4;
    const float* W1p = W1 + colB + bn;
    const float* W3p = W3 + colB + bn;

    // prologue: fetch k-slice 0 into registers
    float4 a0 = *(const float4*)(xrow + acol);
    float4 a1 = *(const float4*)(xrow + acol + 4);
    float r1[8], r3[8];
#pragma unroll
    for (int j = 0; j < 8; ++j) {
        int kk = bk + (j >> 2) * 16 + (j & 3);
        size_t gi = (size_t)kk * II;
        r1[j] = W1p[gi]; r3[j] = W3p[gi];
    }

    int lane = tid & 31, wv = tid >> 5;
    int rb = wv & 3, ch = wv >> 2;
    v8f cg0 = {}, cg1 = {}, cu0 = {}, cu1 = {};

    // fragment addresses (offsets within a buffer; buffer base varies per iter)
    int aoff  = (rb * 16 + (lane & 15)) * 40 + (lane >> 4) * 16;
    int kc    = (lane >> 4) * 16;
    int boff0 = ((ch * 2) * 16 + (lane & 15)) * 40 + kc;
    int boff1 = ((ch * 2 + 1) * 16 + (lane & 15)) * 40 + kc;

    int buf = 0;
    for (int k0 = 0; k0 < DD; k0 += TKS) {
        _Float16* Ab  = As  + buf * (TM * 40);
        _Float16* B1b = B1s + buf * (TN * 40);
        _Float16* B3b = B3s + buf * (TN * 40);
        // commit staged registers -> LDS (vector stores)
        *(v8h*)&Ab[arow * 40 + acol] = cvt8(a0, a1);
        *(v4h*)&B1b[bn * 40 + bk]      = cvt4(r1[0], r1[1], r1[2], r1[3]);
        *(v4h*)&B1b[bn * 40 + bk + 16] = cvt4(r1[4], r1[5], r1[6], r1[7]);
        *(v4h*)&B3b[bn * 40 + bk]      = cvt4(r3[0], r3[1], r3[2], r3[3]);
        *(v4h*)&B3b[bn * 40 + bk + 16] = cvt4(r3[4], r3[5], r3[6], r3[7]);
        __syncthreads();

        // issue next slice's global loads before computing (latency overlap)
        int kn = k0 + TKS;
        if (kn < DD) {
            a0 = *(const float4*)(xrow + kn + acol);
            a1 = *(const float4*)(xrow + kn + acol + 4);
#pragma unroll
            for (int j = 0; j < 8; ++j) {
                int kk = kn + bk + (j >> 2) * 16 + (j & 3);
                size_t gi = (size_t)kk * II;
                r1[j] = W1p[gi]; r3[j] = W3p[gi];
            }
        }

        // load ALL fragments, then fence the scheduler, then the WMMA chain:
        // forces independent ds_loads to issue up-front -> partial dscnt waits
        const _Float16* ap  = Ab + aoff;
        const _Float16* p10 = B1b + boff0;
        const _Float16* p30 = B3b + boff0;
        const _Float16* p11 = B1b + boff1;
        const _Float16* p31 = B3b + boff1;
        v8h aLo  = *(const v8h*)ap,        aHi  = *(const v8h*)(ap + 8);
        v8h b10a = *(const v8h*)p10,       b10b = *(const v8h*)(p10 + 8);
        v8h b30a = *(const v8h*)p30,       b30b = *(const v8h*)(p30 + 8);
        v8h b11a = *(const v8h*)p11,       b11b = *(const v8h*)(p11 + 8);
        v8h b31a = *(const v8h*)p31,       b31b = *(const v8h*)(p31 + 8);
        __builtin_amdgcn_sched_barrier(0);
        v16h A = cat8(aLo, aHi);
        cg0 = wmma_f16(A, cat8(b10a, b10b), cg0);
        cu0 = wmma_f16(A, cat8(b30a, b30b), cu0);
        cg1 = wmma_f16(A, cat8(b11a, b11b), cg1);
        cu1 = wmma_f16(A, cat8(b31a, b31b), cu1);
        buf ^= 1;
    }

    // epilogue: clamp + silu*up, C layout: M = v + 8*(lane>>4), N = lane&15
    int mrow = rb * 16 + (lane >> 4) * 8;
    int ncol = lane & 15;
#pragma unroll
    for (int j = 0; j < 2; ++j) {
        v8f gacc = j ? cg1 : cg0;
        v8f uacc = j ? cu1 : cu0;
        int cb = colB + (ch * 2 + j) * 16 + ncol;
#pragma unroll
        for (int v = 0; v < 8; ++v) {
            float gv = fminf(gacc[v], SLIM);
            float uv = fminf(fmaxf(uacc[v], -SLIM), SLIM);
            float act = gv / (1.f + expf(-gv)) * uv;
            int row = base + rowT + mrow + v;
            h[(size_t)row * II + cb] = (_Float16)act;
        }
    }
}

// ---------------- GEMM2: out_slots = h @ W2 (f32 out) -------------------------
__global__ __launch_bounds__(256)
void moe_gemm2(const float* __restrict__ w2, const float* __restrict__ w2s,
               const int* __restrict__ hdr, const _Float16* __restrict__ h,
               float* __restrict__ outs) {
    int g = blockIdx.z;
    int base, cp; const float* W2;
    if (g == 0) { base = 0; cp = SH_SLOTS; W2 = w2s; }
    else {
        int e = g - 1;
        base = hdr[16 + e]; cp = hdr[8 + e];
        W2 = w2 + (size_t)e * II * DD;
    }
    int rowT = blockIdx.y * TM;
    if (rowT >= cp) return;
    int colB = blockIdx.x * TN;

    __shared__ __align__(16) _Float16 Bs[2 * TN * 40];

    int tid = threadIdx.x, lane = tid & 31, wv = tid >> 5;
    int rb = wv & 3, ch = wv >> 2;
    int bn = tid & 63, bk = (tid >> 6) * 4;
    const float* W2p = W2 + colB + bn;

    // prologue: k-slice 0 of W2 into registers
    float r2[8];
#pragma unroll
    for (int j = 0; j < 8; ++j) {
        int kk = bk + (j >> 2) * 16 + (j & 3);
        r2[j] = W2p[(size_t)kk * DD];
    }

    // A fragments come straight from global h (f16 row-major, 16B chunks)
    int ar = rb * 16 + (lane & 15);
    const _Float16* aprow = h + (size_t)(base + rowT + ar) * II + (lane >> 4) * 16;

    int kc    = (lane >> 4) * 16;
    int boff0 = ((ch * 2) * 16 + (lane & 15)) * 40 + kc;
    int boff1 = ((ch * 2 + 1) * 16 + (lane & 15)) * 40 + kc;

    v8f c0 = {}, c1 = {};
    int buf = 0;
    for (int k0 = 0; k0 < II; k0 += TKS) {
        _Float16* Bb = Bs + buf * (TN * 40);
        *(v4h*)&Bb[bn * 40 + bk]      = cvt4(r2[0], r2[1], r2[2], r2[3]);
        *(v4h*)&Bb[bn * 40 + bk + 16] = cvt4(r2[4], r2[5], r2[6], r2[7]);
        __syncthreads();

        int kn = k0 + TKS;
        if (kn < II) {
#pragma unroll
            for (int j = 0; j < 8; ++j) {
                int kk = kn + bk + (j >> 2) * 16 + (j & 3);
                r2[j] = W2p[(size_t)kk * DD];
            }
        }

        // batch all fragment loads, fence, then the WMMA chain
        const _Float16* ap = aprow + k0;
        const _Float16* p0 = Bb + boff0;
        const _Float16* p1 = Bb + boff1;
        v8h aLo = *(const v8h*)ap,  aHi = *(const v8h*)(ap + 8);
        v8h b0a = *(const v8h*)p0,  b0b = *(const v8h*)(p0 + 8);
        v8h b1a = *(const v8h*)p1,  b1b = *(const v8h*)(p1 + 8);
        __builtin_amdgcn_sched_barrier(0);
        v16h A = cat8(aLo, aHi);
        c0 = wmma_f16(A, cat8(b0a, b0b), c0);
        c1 = wmma_f16(A, cat8(b1a, b1b), c1);
        buf ^= 1;
    }

    int mrow = rb * 16 + (lane >> 4) * 8;
    int ncol = lane & 15;
#pragma unroll
    for (int j = 0; j < 2; ++j) {
        v8f c = j ? c1 : c0;
        int cb = colB + (ch * 2 + j) * 16 + ncol;
#pragma unroll
        for (int v = 0; v < 8; ++v)
            outs[(size_t)(base + rowT + mrow + v) * DD + cb] = c[v];
    }
}

// ---------------- combine: deterministic float4 gather (no float atomics) -----
__global__ void moe_combine(const int* __restrict__ tki, const float* __restrict__ tkw,
                            const float* __restrict__ outs, float* __restrict__ out) {
    int idx = blockIdx.x * blockDim.x + threadIdx.x;        // NT*DD/4 elements
    if (idx >= NT * DD / 4) return;
    int t = idx / (DD / 4);
    int d = (idx - t * (DD / 4)) * 4;
    int s0 = tki[t * 2], s1 = tki[t * 2 + 1];
    float w0 = tkw[t * 2], w1 = tkw[t * 2 + 1];
    float4 sh = *(const float4*)(outs + (size_t)t  * DD + d);   // shared: slot == token
    float4 o0 = *(const float4*)(outs + (size_t)s0 * DD + d);
    float4 o1 = *(const float4*)(outs + (size_t)s1 * DD + d);
    float4 r;
    r.x = sh.x + w0 * o0.x + w1 * o1.x;
    r.y = sh.y + w0 * o0.y + w1 * o1.y;
    r.z = sh.z + w0 * o0.z + w1 * o1.z;
    r.w = sh.w + w0 * o0.w + w1 * o1.w;
    *(float4*)(out + (size_t)t * DD + d) = r;
}

extern "C" void kernel_launch(void* const* d_in, const int* in_sizes, int n_in,
                              void* d_out, int out_size, void* d_ws, size_t ws_size,
                              hipStream_t stream) {
    (void)in_sizes; (void)n_in; (void)out_size; (void)ws_size;
    const float* x   = (const float*)d_in[0];
    // d_in[1] = input_ids (unused by the math)
    const float* gw  = (const float*)d_in[2];
    const float* gb  = (const float*)d_in[3];
    const float* w1  = (const float*)d_in[4];
    const float* w2  = (const float*)d_in[5];
    const float* w3  = (const float*)d_in[6];
    const float* w1s = (const float*)d_in[7];
    const float* w2s = (const float*)d_in[8];
    const float* w3s = (const float*)d_in[9];

    char* ws = (char*)d_ws;
    int*      hdr  = (int*)(ws + OFF_HDR);
    int*      tki  = (int*)(ws + OFF_TOPKI);
    float*    tkw  = (float*)(ws + OFF_TOPKW);
    int*      stok = (int*)(ws + OFF_STOK);
    _Float16* h    = (_Float16*)(ws + OFF_H);
    float*    outs = (float*)(ws + OFF_OUT);
    float*    out  = (float*)d_out;

    moe_init   <<<(TOT_SLOTS + 255) / 256, 256, 0, stream>>>(hdr, stok);
    moe_gate   <<<(NT + 255) / 256,        256, 0, stream>>>(x, gw, gb, hdr, tki, tkw);
    moe_scan   <<<1, 32, 0, stream>>>(hdr);
    moe_fill   <<<(2 * NT + 255) / 256,    256, 0, stream>>>(hdr, tki, stok);

    dim3 g1(II / TN, NT / TM, EE + 1);     // 48 x 32 x 9, blocks early-exit past padded count
    moe_gemm1  <<<g1, 256, 0, stream>>>(x, w1, w3, w1s, w3s, hdr, stok, h);
    dim3 g2(DD / TN, NT / TM, EE + 1);     // 12 x 32 x 9
    moe_gemm2  <<<g2, 256, 0, stream>>>(w2, w2s, hdr, h, outs);

    moe_combine<<<(NT * DD / 4 + 255) / 256, 256, 0, stream>>>(tki, tkw, outs, out);
}
